// GraphSAGEModel_23673859736035
// MI455X (gfx1250) — matrix operand order repaired and checked by reading, not compiled, not run on hardware
//
#include <hip/hip_runtime.h>
#include <hip/hip_bf16.h>
#include <stdint.h>

// ---------------- problem constants (from reference) ----------------
constexpr int kN   = 50000;   // nodes  (== 3125 * 16, exact row tiles)
constexpr int kE   = 800000;  // edges
constexpr int kG   = 64;      // graphs
constexpr int kIN  = 128;     // input dim
constexpr int kH   = 256;     // hidden dim
constexpr int kGD  = 32;      // global dim
constexpr int kMH  = 512;     // mlp hidden
constexpr int kOUT = 10;      // output dim
constexpr float kSLOPE = 0.042859419676898734f;
constexpr float kEPS   = 1e-5f;

constexpr int kMTILES = kN / 16;       // 3125

// ---------------- WMMA types ----------------
typedef __attribute__((ext_vector_type(16))) __bf16 v16bf;
typedef __attribute__((ext_vector_type(8)))  float  v8f;

union Frag {
  uint32_t u[8];
  __bf16   h[16];
  v16bf    v;
};

union Pair {
  uint32_t u;
  __bf16   h[2];
};

__device__ __forceinline__ float leaky(float v) {
  return v > 0.0f ? v : kSLOPE * v;
}

__device__ __forceinline__ void atomAddF(float* p, float v) {
  __hip_atomic_fetch_add(p, v, __ATOMIC_RELAXED, __HIP_MEMORY_SCOPE_AGENT);
}

// Load one 16x32 bf16 A fragment from row-major f32 A[lda], optional per-row scale.
// Layout (ISA 7.12.2): lanes 0-15 row M=lane hold K=0..7,16..23; lanes 16-31 hold K=8..15,24..31.
// f32 -> bf16 via fptrunc cast: lowers to v_cvt_pk_bf16_f32 on gfx1250.
__device__ __forceinline__ void load_a_frag(const float* __restrict__ A, int lda,
                                            int row0, int kbase,
                                            const float* __restrict__ rowScale,
                                            Frag& f) {
  int lane = threadIdx.x & 31;
  int r    = row0 + (lane & 15);
  int k0   = kbase + ((lane < 16) ? 0 : 8);
  const float* p = A + (size_t)r * lda + k0;
  float sc = rowScale ? rowScale[r] : 1.0f;
  float4 lo0 = *(const float4*)(p);
  float4 lo1 = *(const float4*)(p + 4);
  float4 hi0 = *(const float4*)(p + 16);
  float4 hi1 = *(const float4*)(p + 20);
  f.h[0]  = (__bf16)(lo0.x * sc);  f.h[1]  = (__bf16)(lo0.y * sc);
  f.h[2]  = (__bf16)(lo0.z * sc);  f.h[3]  = (__bf16)(lo0.w * sc);
  f.h[4]  = (__bf16)(lo1.x * sc);  f.h[5]  = (__bf16)(lo1.y * sc);
  f.h[6]  = (__bf16)(lo1.z * sc);  f.h[7]  = (__bf16)(lo1.w * sc);
  f.h[8]  = (__bf16)(hi0.x * sc);  f.h[9]  = (__bf16)(hi0.y * sc);
  f.h[10] = (__bf16)(hi0.z * sc);  f.h[11] = (__bf16)(hi0.w * sc);
  f.h[12] = (__bf16)(hi1.x * sc);  f.h[13] = (__bf16)(hi1.y * sc);
  f.h[14] = (__bf16)(hi1.z * sc);  f.h[15] = (__bf16)(hi1.w * sc);
}

// Pre-packed B fragment: 8 contiguous dwords per lane per (kt,nt) tile.
__device__ __forceinline__ void load_b_frag(const uint32_t* __restrict__ P,
                                            int ntiles16, int kt, int nt, Frag& f) {
  const uint32_t* p = P + (((size_t)kt * ntiles16 + nt) * 32 + (threadIdx.x & 31)) * 8;
  *(uint4*)&f.u[0] = *(const uint4*)p;
  *(uint4*)&f.u[4] = *(const uint4*)(p + 4);
}

// ---------------- pack weights into WMMA B-fragment layout ----------------
// W: row-major f32 [K x Nreal], packed to [K/32][Npad/16][lane(32)][dword(8)] bf16 pairs.
__global__ void pack_b_kernel(const float* __restrict__ W, int K, int Nreal, int Npad,
                              uint32_t* __restrict__ P) {
  int ntiles = Npad >> 4;
  int total  = (K >> 5) * ntiles * 256;
  int idx = blockIdx.x * blockDim.x + threadIdx.x;
  if (idx >= total) return;
  int j    = idx & 7;
  int lane = (idx >> 3) & 31;
  int tile = idx >> 8;
  int nt   = tile % ntiles;
  int kt   = tile / ntiles;
  int n    = nt * 16 + (lane & 15);
  int k    = kt * 32 + ((lane < 16) ? 0 : 8) + ((j < 4) ? 2 * j : 8 + 2 * j);
  float a = (n < Nreal) ? W[(size_t)k * Nreal + n] : 0.0f;
  float b = (n < Nreal) ? W[(size_t)(k + 1) * Nreal + n] : 0.0f;
  Pair pr;
  pr.h[0] = (__bf16)a;
  pr.h[1] = (__bf16)b;
  P[idx] = pr.u;
}

// ---------------- utility kernels ----------------
__global__ void zero_kernel(float* __restrict__ p, size_t n) {
  size_t i = (size_t)blockIdx.x * blockDim.x + threadIdx.x;
  size_t stride = (size_t)gridDim.x * blockDim.x;
  for (; i < n; i += stride) p[i] = 0.0f;
}

__global__ void deg_kernel(const int* __restrict__ dst, float* __restrict__ deg) {
  int e = blockIdx.x * blockDim.x + threadIdx.x;
  if (e < kE) atomAddF(&deg[dst[e]], 1.0f);
}

__global__ void invdeg_kernel(float* __restrict__ d) {
  int i = blockIdx.x * blockDim.x + threadIdx.x;
  if (i < kN) d[i] = 1.0f / fmaxf(d[i], 1.0f);
}

// one wave per edge; lane-strided feature atomics (h rows are L2-resident: 51MB < 192MB L2)
__global__ void scatter_kernel(const float* __restrict__ h, const int* __restrict__ src,
                               const int* __restrict__ dst, float* __restrict__ agg, int K) {
  int e = blockIdx.x * (blockDim.x >> 5) + (threadIdx.x >> 5);
  if (e >= kE) return;
  int s = src[e], d = dst[e];
  const float* hp = h + (size_t)s * K;
  float* ap = agg + (size_t)d * K;
  for (int c = threadIdx.x & 31; c < K; c += 32) atomAddF(&ap[c], hp[c]);
}

// ---------------- fused dual-GEMM: Out = (agg*invdeg)@Wl + Hh@Wr + bias ----------------
// One wave computes a 16x128 tile of Out[kN x kH]: 8 accumulators, 16 WMMAs per K-step,
// so each converted A fragment feeds 8 WMMAs.  K is compile-time (128 or 256).
// K-loop unroll capped at 2 to keep live ranges under the 256-VGPR budget (no spills).
template <int K>
__global__ void sage_gemm_kernel(const float* __restrict__ Agg, const float* __restrict__ Hh,
                                 const float* __restrict__ invdeg,
                                 const uint32_t* __restrict__ PWl, const uint32_t* __restrict__ PWr,
                                 const float* __restrict__ bias, float* __restrict__ Out) {
  int mt = blockIdx.x * (blockDim.x >> 5) + (threadIdx.x >> 5);   // wave-uniform
  if (mt >= kMTILES) return;
  int ng   = blockIdx.y;            // 128-col group (kH/128 = 2)
  int lane = threadIdx.x & 31;
  v8f acc[8] = {};
  constexpr int ktiles = K >> 5;
  constexpr int ntiles = kH >> 4;   // 16
#pragma unroll 2
  for (int kt = 0; kt < ktiles; ++kt) {
    Frag aAgg, aH;
    load_a_frag(Agg, K, mt * 16, kt * 32, invdeg, aAgg);
    load_a_frag(Hh,  K, mt * 16, kt * 32, nullptr, aH);
#pragma unroll
    for (int j = 0; j < 8; ++j) {
      Frag bl, br;
      load_b_frag(PWl, ntiles, kt, ng * 8 + j, bl);
      load_b_frag(PWr, ntiles, kt, ng * 8 + j, br);
      acc[j] = __builtin_amdgcn_wmma_f32_16x16x32_bf16(false, aAgg.v, false, bl.v,
                                                       (short)0, acc[j], false, false);
      acc[j] = __builtin_amdgcn_wmma_f32_16x16x32_bf16(false, aH.v, false, br.v,
                                                       (short)0, acc[j], false, false);
    }
  }
  // C layout: VGPR g -> row mt*16 + g + (lane>=16 ? 8 : 0), col = nt*16 + (lane&15)
  int rbase = mt * 16 + ((lane >> 4) << 3);
#pragma unroll
  for (int j = 0; j < 8; ++j) {
    int col  = ng * 128 + j * 16 + (lane & 15);
    float bv = bias[col];
#pragma unroll
    for (int g = 0; g < 8; ++g)
      Out[(size_t)(rbase + g) * kH + col] = acc[j][g] + bv;
  }
}

// ---------------- batch-norm ----------------
__global__ void bn_stats_kernel(const float* __restrict__ Out, float* __restrict__ colsum,
                                float* __restrict__ colsumsq, int rowsPerBlock) {
  int c  = threadIdx.x;             // kH threads
  int r0 = blockIdx.x * rowsPerBlock;
  int r1 = min(r0 + rowsPerBlock, kN);
  float s = 0.0f, s2 = 0.0f;
  for (int r = r0; r < r1; ++r) {
    float v = Out[(size_t)r * kH + c];
    s += v; s2 += v * v;
  }
  atomAddF(&colsum[c], s);
  atomAddF(&colsumsq[c], s2);
}

__global__ void bn_finalize_kernel(const float* __restrict__ colsum,
                                   const float* __restrict__ colsumsq,
                                   const float* __restrict__ gamma,
                                   const float* __restrict__ beta,
                                   float* __restrict__ scale, float* __restrict__ shift) {
  int c = threadIdx.x;
  float mean = colsum[c] * (1.0f / kN);
  float var  = colsumsq[c] * (1.0f / kN) - mean * mean;
  float sc   = gamma[c] * rsqrtf(var + kEPS);
  scale[c] = sc;
  shift[c] = beta[c] - mean * sc;
}

__global__ void bn_apply_kernel(float* __restrict__ h, const float* __restrict__ scale,
                                const float* __restrict__ shift) {
  int c = threadIdx.x;              // kH threads, one row per block
  size_t i = (size_t)blockIdx.x * kH + c;
  h[i] = leaky(h[i] * scale[c] + shift[c]);
}

// ---------------- pooling + concat ----------------
__global__ void pool_kernel(const float* __restrict__ h, const int* __restrict__ batch,
                            float* __restrict__ pooled) {
  int n = blockIdx.x, c = threadIdx.x;
  atomAddF(&pooled[(size_t)batch[n] * kH + c], h[(size_t)n * kH + c]);
}

__global__ void cnt_kernel(const int* __restrict__ batch, float* __restrict__ cnt) {
  int i = blockIdx.x * blockDim.x + threadIdx.x;
  if (i < kN) atomAddF(&cnt[batch[i]], 1.0f);
}

__global__ void build_z_kernel(const float* __restrict__ pooled, const float* __restrict__ cnt,
                               const float* __restrict__ gf, float* __restrict__ z) {
  int i = blockIdx.x * blockDim.x + threadIdx.x;
  constexpr int ZC = kH + kGD;
  if (i >= kG * ZC) return;
  int g = i / ZC, c = i % ZC;
  z[i] = (c < kH) ? pooled[(size_t)g * kH + c] / fmaxf(cnt[g], 1.0f)
                  : gf[(size_t)g * kGD + (c - kH)];
}

// ---------------- generic WMMA GEMM for the MLP head ----------------
template <int ACT>
__global__ void mlp_gemm_kernel(const float* __restrict__ A, int Mtiles, int K,
                                const uint32_t* __restrict__ PB, int NpadTiles16,
                                const float* __restrict__ bias, float* __restrict__ Out,
                                int Nout, int ldout) {
  int mt = blockIdx.x * (blockDim.x >> 5) + (threadIdx.x >> 5);
  if (mt >= Mtiles) return;
  int ng   = blockIdx.y;
  int lane = threadIdx.x & 31;
  v8f acc[4] = {};
  int ktiles = K >> 5;
#pragma unroll 2
  for (int kt = 0; kt < ktiles; ++kt) {
    Frag a;
    load_a_frag(A, K, mt * 16, kt * 32, nullptr, a);
#pragma unroll
    for (int j = 0; j < 4; ++j) {
      Frag b;
      load_b_frag(PB, NpadTiles16, kt, ng * 4 + j, b);
      acc[j] = __builtin_amdgcn_wmma_f32_16x16x32_bf16(false, a.v, false, b.v,
                                                       (short)0, acc[j], false, false);
    }
  }
  int rbase = mt * 16 + ((lane >> 4) << 3);
#pragma unroll
  for (int j = 0; j < 4; ++j) {
    int col = ng * 64 + j * 16 + (lane & 15);
    if (col < Nout) {
      float bv = bias[col];
#pragma unroll
      for (int g = 0; g < 8; ++g) {
        float v = acc[j][g] + bv;
        if (ACT) v = leaky(v);
        Out[(size_t)(rbase + g) * ldout + col] = v;
      }
    }
  }
}

// ---------------- host orchestration ----------------
static inline int cdiv(long long a, long long b) { return (int)((a + b - 1) / b); }
constexpr size_t pksize(int K, int Npad) { return (size_t)(K / 32) * (Npad / 16) * 256; }

extern "C" void kernel_launch(void* const* d_in, const int* in_sizes, int n_in,
                              void* d_out, int out_size, void* d_ws, size_t ws_size,
                              hipStream_t stream) {
  (void)in_sizes; (void)n_in; (void)out_size; (void)ws_size;
  const float* x     = (const float*)d_in[0];
  const int*   eidx  = (const int*)d_in[1];
  const int*   srcI  = eidx;
  const int*   dstI  = eidx + kE;
  const int*   batch = (const int*)d_in[2];
  const float* gf    = (const float*)d_in[3];
  const float* Wl0   = (const float*)d_in[4];
  const float* Wr0   = (const float*)d_in[5];
  const float* bl0   = (const float*)d_in[6];
  const float* Wl    = (const float*)d_in[7];   // [2, kH, kH]
  const float* Wr    = (const float*)d_in[8];
  const float* bl    = (const float*)d_in[9];   // [2, kH]
  const float* bng   = (const float*)d_in[10];  // [3, kH]
  const float* bnb   = (const float*)d_in[11];
  const float* W1    = (const float*)d_in[12];
  const float* b1    = (const float*)d_in[13];
  const float* W2    = (const float*)d_in[14];
  const float* b2    = (const float*)d_in[15];
  const float* W3    = (const float*)d_in[16];
  const float* b3    = (const float*)d_in[17];
  float* out = (float*)d_out;

  // ---- workspace carve ----
  float* wsf = (float*)d_ws;
  size_t off = 0;
  float* hA       = wsf + off; off += (size_t)kN * kH;
  float* hB       = wsf + off; off += (size_t)kN * kH;
  float* agg      = wsf + off; off += (size_t)kN * kH;
  float* invdeg   = wsf + off; off += kN;
  float* colsum   = wsf + off; off += kH;
  float* colsumsq = wsf + off; off += kH;
  float* bnscale  = wsf + off; off += kH;
  float* bnshift  = wsf + off; off += kH;
  float* pooled   = wsf + off; off += (size_t)kG * kH;
  float* cnt      = wsf + off; off += kG;
  float* z        = wsf + off; off += (size_t)kG * (kH + kGD);
  float* z1       = wsf + off; off += (size_t)kG * kMH;
  float* z2       = wsf + off; off += (size_t)kG * kMH;
  uint32_t* pk = (uint32_t*)(wsf + off);
  uint32_t* PWl0 = pk;                       pk += pksize(kIN, kH);
  uint32_t* PWr0 = pk;                       pk += pksize(kIN, kH);
  uint32_t* PWl1 = pk;                       pk += pksize(kH, kH);
  uint32_t* PWr1 = pk;                       pk += pksize(kH, kH);
  uint32_t* PWl2 = pk;                       pk += pksize(kH, kH);
  uint32_t* PWr2 = pk;                       pk += pksize(kH, kH);
  uint32_t* PW1  = pk;                       pk += pksize(kH + kGD, kMH);
  uint32_t* PW2  = pk;                       pk += pksize(kMH, kMH);
  uint32_t* PW3  = pk;                       pk += pksize(kMH, 64);

  // ---- pack weights to bf16 WMMA B-fragment layout ----
  auto pack = [&](const float* W, int K, int Nreal, int Npad, uint32_t* P) {
    int total = (K / 32) * (Npad / 16) * 256;
    pack_b_kernel<<<cdiv(total, 256), 256, 0, stream>>>(W, K, Nreal, Npad, P);
  };
  pack(Wl0, kIN, kH, kH, PWl0);
  pack(Wr0, kIN, kH, kH, PWr0);
  pack(Wl,                   kH, kH, kH, PWl1);
  pack(Wl + (size_t)kH * kH, kH, kH, kH, PWl2);
  pack(Wr,                   kH, kH, kH, PWr1);
  pack(Wr + (size_t)kH * kH, kH, kH, kH, PWr2);
  pack(W1, kH + kGD, kMH, kMH, PW1);
  pack(W2, kMH, kMH, kMH, PW2);
  pack(W3, kMH, kOUT, 64, PW3);

  // ---- degrees ----
  zero_kernel<<<256, 256, 0, stream>>>(invdeg, kN);
  deg_kernel<<<cdiv(kE, 256), 256, 0, stream>>>(dstI, invdeg);
  invdeg_kernel<<<cdiv(kN, 256), 256, 0, stream>>>(invdeg);

  // ---- SAGE layers ----
  auto sage_prolog = [&](const float* hcur, int K) {
    zero_kernel<<<1024, 256, 0, stream>>>(agg, (size_t)kN * K);
    scatter_kernel<<<cdiv(kE, 8), 256, 0, stream>>>(hcur, srcI, dstI, agg, K);
  };
  auto sage_epilog = [&](float* hnext, const float* gamma, const float* beta) {
    zero_kernel<<<2, 256, 0, stream>>>(colsum, 2 * kH);   // colsum + colsumsq (contiguous)
    bn_stats_kernel<<<cdiv(kN, 250), kH, 0, stream>>>(hnext, colsum, colsumsq, 250);
    bn_finalize_kernel<<<1, kH, 0, stream>>>(colsum, colsumsq, gamma, beta, bnscale, bnshift);
    bn_apply_kernel<<<kN, kH, 0, stream>>>(hnext, bnscale, bnshift);
  };
  dim3 ggrid(cdiv(kMTILES, 8), kH / 128);

  sage_prolog(x, kIN);
  sage_gemm_kernel<kIN><<<ggrid, 256, 0, stream>>>(agg, x, invdeg, PWl0, PWr0, bl0, hA);
  sage_epilog(hA, bng, bnb);

  sage_prolog(hA, kH);
  sage_gemm_kernel<kH><<<ggrid, 256, 0, stream>>>(agg, hA, invdeg, PWl1, PWr1, bl, hB);
  sage_epilog(hB, bng + kH, bnb + kH);

  sage_prolog(hB, kH);
  sage_gemm_kernel<kH><<<ggrid, 256, 0, stream>>>(agg, hB, invdeg, PWl2, PWr2, bl + kH, hA);
  sage_epilog(hA, bng + 2 * kH, bnb + 2 * kH);

  // ---- global mean pool + concat ----
  zero_kernel<<<64, 256, 0, stream>>>(pooled, (size_t)kG * kH + kG);  // pooled + cnt (contiguous)
  pool_kernel<<<kN, kH, 0, stream>>>(hA, batch, pooled);
  cnt_kernel<<<cdiv(kN, 256), 256, 0, stream>>>(batch, cnt);
  build_z_kernel<<<cdiv(kG * (kH + kGD), 256), 256, 0, stream>>>(pooled, cnt, gf, z);

  // ---- MLP head (G=64 rows -> 4 row tiles) ----
  {
    dim3 g1(1, kMH / 64);
    mlp_gemm_kernel<1><<<g1, 128, 0, stream>>>(z, kG / 16, kH + kGD, PW1, kMH / 16, b1, z1, kMH, kMH);
    dim3 g2(1, kMH / 64);
    mlp_gemm_kernel<1><<<g2, 128, 0, stream>>>(z1, kG / 16, kMH, PW2, kMH / 16, b2, z2, kMH, kMH);
    dim3 g3(1, 1);
    mlp_gemm_kernel<0><<<g3, 128, 0, stream>>>(z2, kG / 16, kMH, PW3, 64 / 16, b3, out, kOUT, kOUT);
  }
}
